// JointModelV4_89781996356243
// MI455X (gfx1250) — matrix operand order repaired and checked by reading, not compiled, not run on hardware
//
#include <hip/hip_runtime.h>
#include <math.h>

typedef __attribute__((ext_vector_type(16))) _Float16 v16h;
typedef __attribute__((ext_vector_type(8)))  _Float16 v8h;
typedef __attribute__((ext_vector_type(8)))  float    v8f;
typedef unsigned int __attribute__((ext_vector_type(4))) u32x4;
typedef int          __attribute__((ext_vector_type(4))) i32x4;
typedef int          __attribute__((ext_vector_type(8))) i32x8;

#if __has_builtin(__builtin_amdgcn_tensor_load_to_lds) && \
    __has_builtin(__builtin_amdgcn_s_wait_tensorcnt)
#define HAVE_TDM 1
#endif

#define B_    2048
#define L_    16
#define N_    4
#define K_    8
#define D_    256
#define D4_   64
#define NHEAD_ 4
#define ROWS_ (B_ * L_ * K_)   /* 262144 */
#define BK_   (B_ * K_)        /* 16384  */
#define NOISE_ (1.0 + 1e-6)
#define COMBW_ 288             /* 262 padded to multiple of 32 */

// ---------------------------------------------------------------------------
// complex double helpers (reference does MMSE in complex128)
// ---------------------------------------------------------------------------
struct cplx { double re, im; };
__device__ __forceinline__ cplx cadd(cplx a, cplx b) { return {a.re + b.re, a.im + b.im}; }
__device__ __forceinline__ cplx csub(cplx a, cplx b) { return {a.re - b.re, a.im - b.im}; }
__device__ __forceinline__ cplx cmul(cplx a, cplx b) {
  return {a.re * b.re - a.im * b.im, a.re * b.im + a.im * b.re};
}
__device__ __forceinline__ cplx cmulc(cplx a, cplx b) {  // conj(a) * b
  return {a.re * b.re + a.im * b.im, a.re * b.im - a.im * b.re};
}

// ---------------------------------------------------------------------------
// Kernel 1: global MMSE  (one thread per batch; K=8 complex solve)
// ---------------------------------------------------------------------------
__global__ void global_mmse_kernel(const float* __restrict__ Hq,
                                   const float* __restrict__ yq,
                                   float* __restrict__ sglob /* BK*2 */) {
  int b = blockIdx.x * blockDim.x + threadIdx.x;
  if (b >= B_) return;
  cplx R[K_][K_]; cplx rhs[K_];
  for (int i = 0; i < K_; ++i) { rhs[i] = {0, 0}; for (int j = 0; j < K_; ++j) R[i][j] = {0, 0}; }
  for (int m = 0; m < L_ * N_; ++m) {
    cplx h[K_];
    const float* hp = Hq + ((size_t)b * L_ * N_ + m) * K_ * 2;
    for (int k = 0; k < K_; ++k) h[k] = {(double)hp[k * 2], (double)hp[k * 2 + 1]};
    const float* yp = yq + ((size_t)b * L_ * N_ + m) * 2;
    cplx y = {(double)yp[0], (double)yp[1]};
    for (int i = 0; i < K_; ++i) {
      for (int j = 0; j < K_; ++j) R[i][j] = cadd(R[i][j], cmulc(h[i], h[j]));
      rhs[i] = cadd(rhs[i], cmulc(h[i], y));
    }
  }
  for (int i = 0; i < K_; ++i) R[i][i].re += NOISE_;
  for (int p = 0; p < K_; ++p) {
    cplx piv = R[p][p];
    double d = piv.re * piv.re + piv.im * piv.im;
    cplx ip = {piv.re / d, -piv.im / d};
    for (int j = 0; j < K_; ++j) R[p][j] = cmul(R[p][j], ip);
    rhs[p] = cmul(rhs[p], ip);
    for (int r = 0; r < K_; ++r) {
      if (r == p) continue;
      cplx f = R[r][p];
      for (int j = 0; j < K_; ++j) R[r][j] = csub(R[r][j], cmul(f, R[p][j]));
      rhs[r] = csub(rhs[r], cmul(f, rhs[p]));
    }
  }
  for (int k = 0; k < K_; ++k) {
    sglob[((size_t)b * K_ + k) * 2 + 0] = (float)rhs[k].re;
    sglob[((size_t)b * K_ + k) * 2 + 1] = (float)rhs[k].im;
  }
}

// ---------------------------------------------------------------------------
// Kernel 2: per-AP MMSE  (one thread per (b,l); 4x4 complex inverse)
// ---------------------------------------------------------------------------
__global__ void per_ap_mmse_kernel(const float* __restrict__ Hq,
                                   const float* __restrict__ yq,
                                   float* __restrict__ s_init /* ROWS*2 */) {
  int t = blockIdx.x * blockDim.x + threadIdx.x;
  if (t >= B_ * L_) return;
  cplx H[N_][K_];
  const float* hp = Hq + (size_t)t * N_ * K_ * 2;
  for (int n = 0; n < N_; ++n)
    for (int k = 0; k < K_; ++k)
      H[n][k] = {(double)hp[(n * K_ + k) * 2], (double)hp[(n * K_ + k) * 2 + 1]};
  cplx Ry[N_][N_], Mi[N_][N_];
  for (int i = 0; i < N_; ++i)
    for (int j = 0; j < N_; ++j) {
      cplx s = {0, 0};
      for (int k = 0; k < K_; ++k) s = cadd(s, cmulc(H[j][k], H[i][k]));
      Ry[i][j] = s;
      Mi[i][j] = {(i == j) ? 1.0 : 0.0, 0.0};
    }
  for (int i = 0; i < N_; ++i) Ry[i][i].re += NOISE_;
  for (int p = 0; p < N_; ++p) {
    cplx piv = Ry[p][p];
    double d = piv.re * piv.re + piv.im * piv.im;
    cplx ip = {piv.re / d, -piv.im / d};
    for (int j = 0; j < N_; ++j) { Ry[p][j] = cmul(Ry[p][j], ip); Mi[p][j] = cmul(Mi[p][j], ip); }
    for (int r = 0; r < N_; ++r) {
      if (r == p) continue;
      cplx f = Ry[r][p];
      for (int j = 0; j < N_; ++j) {
        Ry[r][j] = csub(Ry[r][j], cmul(f, Ry[p][j]));
        Mi[r][j] = csub(Mi[r][j], cmul(f, Mi[p][j]));
      }
    }
  }
  const float* yp = yq + (size_t)t * N_ * 2;
  cplx y[N_];
  for (int n = 0; n < N_; ++n) y[n] = {(double)yp[n * 2], (double)yp[n * 2 + 1]};
  cplx z[N_];
  for (int m = 0; m < N_; ++m) {
    cplx s = {0, 0};
    for (int n = 0; n < N_; ++n) s = cadd(s, cmul(Mi[m][n], y[n]));
    z[m] = s;
  }
  for (int k = 0; k < K_; ++k) {
    cplx s = {0, 0};
    for (int m = 0; m < N_; ++m) s = cadd(s, cmulc(H[m][k], z[m]));
    s_init[((size_t)t * K_ + k) * 2 + 0] = (float)s.re;
    s_init[((size_t)t * K_ + k) * 2 + 1] = (float)s.im;
  }
}

// ---------------------------------------------------------------------------
// Kernel 3: per-node feature fusion input (4 small MLP2 branches + tail)
// ---------------------------------------------------------------------------
struct FeatArgs {
  const float *s_init, *s_hat, *Hq, *sglob, *bitw, *lsnr;
  const float *iW1, *ib1, *iW2, *ib2;
  const float *dW1, *db1, *dW2, *db2;
  const float *gW1, *gb1, *gW2, *gb2;
  const float *cW1, *cb1, *cW2, *cb2;
  _Float16* comb;
};

__global__ void features_kernel(FeatArgs fa) {
  __shared__ float sh[D4_];
  __shared__ float hpow[K_];
  int row = blockIdx.x;
  int j = threadIdx.x;
  int b = row / (L_ * K_);
  int l = (row / K_) % L_;
  int k = row % K_;
  _Float16* out = fa.comb + (size_t)row * COMBW_;

  if (j < K_) {
    float s = 0.f;
    for (int n = 0; n < N_; ++n) {
      size_t idx = ((((size_t)b * L_ + l) * N_ + n) * K_ + j) * 2;
      float re = fa.Hq[idx], im = fa.Hq[idx + 1];
      s += re * re + im * im;
    }
    hpow[j] = s;
  }
  __syncthreads();

  float x[8];
  x[0] = fa.s_init[(size_t)row * 2]; x[1] = fa.s_init[(size_t)row * 2 + 1];
  {
    float a = fa.ib1[j];
    for (int i = 0; i < 2; ++i) a += x[i] * fa.iW1[i * D4_ + j];
    sh[j] = fmaxf(a, 0.f); __syncthreads();
    float o = fa.ib2[j];
    for (int i = 0; i < D4_; ++i) o += sh[i] * fa.iW2[i * D4_ + j];
    __syncthreads();
    out[0 * D4_ + j] = (_Float16)o;
  }
  x[0] = fa.s_hat[(size_t)row * 2]; x[1] = fa.s_hat[(size_t)row * 2 + 1];
  {
    float a = fa.db1[j];
    for (int i = 0; i < 2; ++i) a += x[i] * fa.dW1[i * D4_ + j];
    sh[j] = fmaxf(a, 0.f); __syncthreads();
    float o = fa.db2[j];
    for (int i = 0; i < D4_; ++i) o += sh[i] * fa.dW2[i * D4_ + j];
    __syncthreads();
    out[1 * D4_ + j] = (_Float16)o;
  }
  x[0] = fa.sglob[((size_t)b * K_ + k) * 2]; x[1] = fa.sglob[((size_t)b * K_ + k) * 2 + 1];
  {
    float a = fa.gb1[j];
    for (int i = 0; i < 2; ++i) a += x[i] * fa.gW1[i * D4_ + j];
    sh[j] = fmaxf(a, 0.f); __syncthreads();
    float o = fa.gb2[j];
    for (int i = 0; i < D4_; ++i) o += sh[i] * fa.gW2[i * D4_ + j];
    __syncthreads();
    out[2 * D4_ + j] = (_Float16)o;
  }
  for (int n = 0; n < N_; ++n)
    for (int c = 0; c < 2; ++c)
      x[n * 2 + c] = fa.Hq[((((size_t)b * L_ + l) * N_ + n) * K_ + k) * 2 + c];
  {
    float a = fa.cb1[j];
    for (int i = 0; i < 2 * N_; ++i) a += x[i] * fa.cW1[i * D4_ + j];
    sh[j] = fmaxf(a, 0.f); __syncthreads();
    float o = fa.cb2[j];
    for (int i = 0; i < D4_; ++i) o += sh[i] * fa.cW2[i * D4_ + j];
    __syncthreads();
    out[3 * D4_ + j] = (_Float16)o;
  }
  if (j < 32) {
    float v = 0.f;
    if (j < 3)       v = fa.bitw[(size_t)row * 3 + j];
    else if (j == 3) v = log1pf(hpow[k]);
    else if (j == 4) {
      float s = 0.f;
      for (int i = 0; i < K_; ++i) s += hpow[i];
      v = log1pf(s - hpow[k]);
    }
    else if (j == 5) v = fa.lsnr[row];
    out[256 + j] = (_Float16)v;
  }
}

// ---------------------------------------------------------------------------
// WMMA GEMM:  C[M,N] = A[M,Kd](f16) * W[Kd,N](f16) + bias, optional ReLU
// block = 128 threads (4 waves); block tile = 64(M) x 32(N); each wave keeps
// two 16x16 accumulators sharing one A fragment.
//
// B-tile staging: wave 0 issues a TDM tensor_load_to_lds with a 2D D#
// (32x32 f16 tile of W, dim0_stride = ldw) into a row-major staging buffer,
// waits with s_wait_tensorcnt, then all threads transpose it to [n][k] in LDS
// so per-lane WMMA fragments are contiguous ds_load_b128. A fragments are two
// contiguous global_load_b128 per lane (ISA 16-bit A 16x32 layout).
// ---------------------------------------------------------------------------
__global__ void gemm_wmma_kernel(const _Float16* __restrict__ A, int lda,
                                 const _Float16* __restrict__ W, int ldw,
                                 const float* __restrict__ bias,
                                 float* __restrict__ C32, int ldc32,
                                 _Float16* __restrict__ C16, int ldc16,
                                 int M, int Kd, int Nd, int relu) {
#ifdef HAVE_TDM
  __shared__ _Float16 Stage[32 * 32];        // row-major [k][n] TDM destination
#endif
  __shared__ _Float16 Bsh[32 * 32];          // transposed: [n][k], n,k in 0..31
  int lane  = threadIdx.x & 31;
  int wv    = threadIdx.x >> 5;
  int mBase = blockIdx.y * 64 + wv * 16;
  int nBase = blockIdx.x * 32;
  int mrow  = mBase + (lane & 15);
  int half  = lane >> 4;                     // 0 or 1 (half-wave)
  int aHi   = half ? 8 : 0;                  // A K-offset (halves)
  int bHi   = half ? 8 : 0;                  // B K-offset (halves)
  int nSub  = lane & 15;
  const _Float16* Arow = A + (size_t)mrow * lda;
  int kk = threadIdx.x >> 2;                 // B-tile row this thread handles
  int ch = (threadIdx.x & 3) * 8;            // 8-half chunk within row
  v8f acc0 = {}, acc1 = {};
#pragma unroll 2
  for (int k0 = 0; k0 < Kd; k0 += 32) {
#ifdef HAVE_TDM
    // --- TDM: async DMA of the 32x32 B tile into LDS (wave 0 only) ---
    if (wv == 0) {
      unsigned long long ga =
          (unsigned long long)(const void*)(W + (size_t)k0 * ldw + nBase);
      unsigned ldsb = (unsigned)(size_t)(void*)Stage;
      u32x4 g0 = { 1u,                                   // count=1 valid D#
                   ldsb,                                  // lds_addr
                   (unsigned)(ga & 0xffffffffu),          // global_addr[31:0]
                   (unsigned)((ga >> 32) & 0x01ffffffu)   // global_addr[56:32]
                     | (2u << 30) };                      // type=2 (image)
      i32x8 g1 = { (int)(1u << 16),                       // data_size=1 -> 2B
                   (int)((Nd & 0xffff) << 16),            // tensor_dim0 lo
                   (int)(((unsigned)Nd >> 16) | ((Kd & 0xffff) << 16)),
                   (int)(((unsigned)Kd >> 16) | (32u << 16)),  // tile_dim0=32
                   32,                                    // tile_dim1=32
                   ldw,                                   // dim0_stride lo
                   0, 0 };
      i32x4 gz = { 0, 0, 0, 0 };
#if defined(__clang_major__) && __clang_major__ >= 23
      i32x8 gz8 = { 0, 0, 0, 0, 0, 0, 0, 0 };
      __builtin_amdgcn_tensor_load_to_lds(g0, g1, gz, gz, gz8, 0);
#else
      __builtin_amdgcn_tensor_load_to_lds(g0, g1, gz, gz, 0);
#endif
      __builtin_amdgcn_s_wait_tensorcnt(0);
    }
    __syncthreads();
    // transpose staged tile -> Bsh[n][k] (wide LDS read, scatter store)
    v8h w = *(const v8h*)(Stage + kk * 32 + ch);
#else
    // fallback: direct global load of B tile chunk
    v8h w = *(const v8h*)(W + (size_t)(k0 + kk) * ldw + nBase + ch);
#endif
#pragma unroll
    for (int e = 0; e < 8; ++e) Bsh[(ch + e) * 32 + kk] = w[e];
    if (k0 + 32 < Kd) __builtin_prefetch(Arow + k0 + 32, 0, 1);
    __syncthreads();
    // A fragment: two contiguous 16B runs (ISA 16-bit A 16x32 layout)
    union { v16h v; v8h h[2]; } au, b0u, b1u;
    au.h[0] = *(const v8h*)(Arow + k0 + aHi);
    au.h[1] = *(const v8h*)(Arow + k0 + aHi + 16);
    // B fragments: two 8-half runs at k=bHi and bHi+16 in transposed LDS (ds_load_b128 x2)
    const _Float16* bp0 = &Bsh[nSub * 32 + bHi];
    const _Float16* bp1 = &Bsh[(nSub + 16) * 32 + bHi];
    b0u.h[0] = *(const v8h*)bp0; b0u.h[1] = *(const v8h*)(bp0 + 16);
    b1u.h[0] = *(const v8h*)bp1; b1u.h[1] = *(const v8h*)(bp1 + 16);
    acc0 = __builtin_amdgcn_wmma_f32_16x16x32_f16(
        false, au.v, false, b0u.v, (short)0, acc0, false, false);
    acc1 = __builtin_amdgcn_wmma_f32_16x16x32_f16(
        false, au.v, false, b1u.v, (short)0, acc1, false, false);
    __syncthreads();
  }
  // epilogue: C/D layout -> VGPR r holds M = mBase + r (+8 for upper half-wave)
  int mOff = half ? 8 : 0;
  for (int tile = 0; tile < 2; ++tile) {
    v8f acc = tile ? acc1 : acc0;
    int nCol = nBase + tile * 16 + nSub;
    float bv = bias ? bias[nCol] : 0.f;
#pragma unroll
    for (int r = 0; r < 8; ++r) {
      int m = mBase + r + mOff;
      float v = acc[r] + bv;
      if (relu) v = fmaxf(v, 0.f);
      if (C32) C32[(size_t)m * ldc32 + nCol] = v;
      if (C16) C16[(size_t)m * ldc16 + nCol] = (_Float16)v;
    }
  }
}

// ---------------------------------------------------------------------------
// deterministic BatchNorm (two-stage) + apply
// ---------------------------------------------------------------------------
__global__ void bn_stat1_kernel(const float* __restrict__ X,
                                float* __restrict__ psum, float* __restrict__ psq) {
  int j = threadIdx.x, blk = blockIdx.x;
  size_t base = (size_t)blk * 128 * D_;
  float s = 0.f, s2 = 0.f;
  for (int r = 0; r < 128; ++r) {
    float v = X[base + (size_t)r * D_ + j];
    s += v; s2 += v * v;
  }
  psum[(size_t)blk * D_ + j] = s;
  psq [(size_t)blk * D_ + j] = s2;
}

__global__ void bn_stat2_kernel(const float* __restrict__ psum, const float* __restrict__ psq,
                                float* __restrict__ stats /*512: mu|var*/, int nblk) {
  int j = threadIdx.x;
  float s = 0.f, s2 = 0.f;
  for (int b = 0; b < nblk; ++b) { s += psum[(size_t)b * D_ + j]; s2 += psq[(size_t)b * D_ + j]; }
  float mu = s / (float)ROWS_;
  stats[j] = mu;
  stats[D_ + j] = s2 / (float)ROWS_ - mu * mu;
}

__global__ void bn_apply_kernel(const float* __restrict__ X, const float* __restrict__ stats,
                                const float* __restrict__ g, const float* __restrict__ b,
                                float* __restrict__ Y32, _Float16* __restrict__ Y16, size_t n) {
  for (size_t i = (size_t)blockIdx.x * blockDim.x + threadIdx.x; i < n;
       i += (size_t)gridDim.x * blockDim.x) {
    int c = (int)(i & (D_ - 1));
    float v = (X[i] - stats[c]) * rsqrtf(stats[D_ + c] + 1e-5f) * g[c] + b[c];
    Y32[i] = v; Y16[i] = (_Float16)v;
  }
}

// ---------------------------------------------------------------------------
// GNN support kernels
// ---------------------------------------------------------------------------
__global__ void mean_msg_kernel(const float* __restrict__ msg, float* __restrict__ mm) {
  size_t i = (size_t)blockIdx.x * blockDim.x + threadIdx.x;
  if (i >= (size_t)BK_ * D_) return;
  int d = (int)(i & (D_ - 1));
  int bk = (int)(i >> 8);
  int b = bk / K_, k = bk % K_;
  float s = 0.f;
  for (int l = 0; l < L_; ++l)
    s += msg[((((size_t)b * L_ + l) * K_ + k) << 8) + d];
  mm[i] = s * (1.f / L_);
}

__global__ void concat_upd_kernel(const _Float16* __restrict__ h16,
                                  const float* __restrict__ mm,
                                  _Float16* __restrict__ X /*ROWS*512*/) {
  size_t i = (size_t)blockIdx.x * blockDim.x + threadIdx.x;
  if (i >= (size_t)ROWS_ * D_) return;
  int d = (int)(i & (D_ - 1));
  size_t row = i >> 8;
  int b = (int)(row / (L_ * K_));
  int k = (int)(row % K_);
  X[row * 512 + d]       = h16[i];
  X[row * 512 + 256 + d] = (_Float16)mm[(((size_t)b * K_ + k) << 8) + d];
}

__global__ void ln_kernel(const float* __restrict__ X, const float* __restrict__ R,
                          const float* __restrict__ g, const float* __restrict__ b,
                          float* __restrict__ Y32, _Float16* __restrict__ Y16) {
  __shared__ float red[D_];
  size_t row = blockIdx.x;
  int j = threadIdx.x;
  size_t idx = row * D_ + j;
  float v = X[idx] + (R ? R[idx] : 0.f);
  red[j] = v; __syncthreads();
  for (int s = 128; s > 0; s >>= 1) { if (j < s) red[j] += red[j + s]; __syncthreads(); }
  float mu = red[0] * (1.f / D_); __syncthreads();
  float dv = v - mu;
  red[j] = dv * dv; __syncthreads();
  for (int s = 128; s > 0; s >>= 1) { if (j < s) red[j] += red[j + s]; __syncthreads(); }
  float o = dv * rsqrtf(red[0] * (1.f / D_) + 1e-5f) * g[j] + b[j];
  if (Y32) Y32[idx] = o;
  if (Y16) Y16[idx] = (_Float16)o;
}

__global__ void agg_logit_kernel(const _Float16* __restrict__ hid, const float* __restrict__ W2,
                                 const float* __restrict__ b2, float* __restrict__ logit) {
  size_t row = (size_t)blockIdx.x * blockDim.x + threadIdx.x;
  if (row >= ROWS_) return;
  float s = b2[0];
  const _Float16* hp = hid + row * 128;
  for (int i = 0; i < 128; ++i) s += (float)hp[i] * W2[i];
  logit[row] = s;
}

__global__ void attn_pool_kernel(const float* __restrict__ h, const float* __restrict__ logit,
                                 float* __restrict__ u32, _Float16* __restrict__ u16) {
  __shared__ float w[L_];
  int bk = blockIdx.x;
  int b = bk / K_, k = bk % K_;
  int j = threadIdx.x;
  if (j < L_) w[j] = logit[((size_t)b * L_ + j) * K_ + k];
  __syncthreads();
  if (j == 0) {
    float mx = w[0];
    for (int l = 1; l < L_; ++l) mx = fmaxf(mx, w[l]);
    float s = 0.f;
    for (int l = 0; l < L_; ++l) { w[l] = expf(w[l] - mx); s += w[l]; }
    float inv = 1.f / s;
    for (int l = 0; l < L_; ++l) w[l] *= inv;
  }
  __syncthreads();
  float acc = 0.f;
  for (int l = 0; l < L_; ++l)
    acc += h[((((size_t)b * L_ + l) * K_ + k) << 8) + j] * w[l];
  u32[((size_t)bk << 8) + j] = acc;
  u16[((size_t)bk << 8) + j] = (_Float16)acc;
}

__global__ void mha_attn_kernel(const float* __restrict__ Q, const float* __restrict__ Kt,
                                const float* __restrict__ V, _Float16* __restrict__ O16) {
  int t = blockIdx.x * blockDim.x + threadIdx.x;
  if (t >= B_ * NHEAD_ * K_) return;
  int qpos = t % K_;
  int head = (t / K_) % NHEAD_;
  int b    = t / (K_ * NHEAD_);
  const int dh = D_ / NHEAD_;
  size_t qoff = ((size_t)b * K_ + qpos) * D_ + head * dh;
  float sc[K_], mx = -3.4e38f;
  for (int kp = 0; kp < K_; ++kp) {
    size_t koff = ((size_t)b * K_ + kp) * D_ + head * dh;
    float s = 0.f;
    for (int d = 0; d < dh; ++d) s += Q[qoff + d] * Kt[koff + d];
    s *= 0.125f;
    sc[kp] = s;
    mx = fmaxf(mx, s);
  }
  float den = 0.f;
  for (int kp = 0; kp < K_; ++kp) { sc[kp] = expf(sc[kp] - mx); den += sc[kp]; }
  float inv = 1.f / den;
  for (int d = 0; d < dh; ++d) {
    float o = 0.f;
    for (int kp = 0; kp < K_; ++kp)
      o += sc[kp] * V[((size_t)b * K_ + kp) * D_ + head * dh + d];
    O16[qoff + d] = (_Float16)(o * inv);
  }
}

__global__ void add_kernel(float* __restrict__ X, const float* __restrict__ Y, size_t n) {
  for (size_t i = (size_t)blockIdx.x * blockDim.x + threadIdx.x; i < n;
       i += (size_t)gridDim.x * blockDim.x)
    X[i] += Y[i];
}

__global__ void final_kernel(const float* __restrict__ user, const float* __restrict__ Wout,
                             const float* __restrict__ bout, const float* __restrict__ alpha,
                             const float* __restrict__ sglob, float* __restrict__ det) {
  int t = blockIdx.x * blockDim.x + threadIdx.x;
  if (t >= BK_) return;
  float r0 = bout[0], r1 = bout[1];
  const float* up = user + (size_t)t * D_;
  for (int i = 0; i < D_; ++i) {
    float u = up[i];
    r0 += u * Wout[i * 2 + 0];
    r1 += u * Wout[i * 2 + 1];
  }
  float a = alpha[0];
  det[(size_t)t * 2 + 0] = sglob[(size_t)t * 2 + 0] + a * r0;
  det[(size_t)t * 2 + 1] = sglob[(size_t)t * 2 + 1] + a * r1;
}

__global__ void cvt16_kernel(const float* __restrict__ src, _Float16* __restrict__ dst,
                             int srcRows, int cols, size_t n) {
  for (size_t i = (size_t)blockIdx.x * blockDim.x + threadIdx.x; i < n;
       i += (size_t)gridDim.x * blockDim.x) {
    size_t r = i / (size_t)cols;
    dst[i] = (r < (size_t)srcRows) ? (_Float16)src[i] : (_Float16)0.f;
  }
}

// ===========================================================================
// host driver
// ===========================================================================
extern "C" void kernel_launch(void* const* d_in, const int* in_sizes, int n_in,
                              void* d_out, int out_size, void* d_ws, size_t ws_size,
                              hipStream_t stream) {
  (void)in_sizes; (void)n_in; (void)out_size; (void)ws_size;
  auto F = [&](int i) { return (const float*)d_in[i]; };

  const float* s_hat = F(0);
  const float* Hq    = F(1);
  const float* yq    = F(2);
  const float* bitw  = F(3);
  const float* lsnr  = F(4);
  const float* alpha = F(5);
  const float *iW1 = F(6),  *ib1 = F(7),  *iW2 = F(8),  *ib2 = F(9);
  const float *dW1 = F(10), *db1 = F(11), *dW2 = F(12), *db2 = F(13);
  const float *gW1 = F(14), *gb1 = F(15), *gW2 = F(16), *gb2 = F(17);
  const float *cW1 = F(18), *cb1 = F(19), *cW2 = F(20), *cb2 = F(21);
  const float *fW1 = F(22), *fb1 = F(23), *fW2 = F(24), *fb2 = F(25);
  const float *bng = F(26), *bnb = F(27);
  int gbase[3] = {28, 38, 48};
  const float *aW1 = F(58), *ab1 = F(59), *aW2 = F(60), *ab2 = F(61);
  int tbase[2] = {62, 78};
  const float *oW = F(94), *ob = F(95);

  float* fout = (float*)d_out;
  float* det   = fout;
  float* sglob = fout + BK_ * 2;

  char* wp = (char*)d_ws;
  auto alloc = [&](size_t bytes) -> void* {
    void* p = (void*)wp;
    wp += (bytes + 255) & ~(size_t)255;
    return p;
  };
  float*    s_init = (float*)alloc((size_t)ROWS_ * 2 * 4);
  _Float16* t2_16  = (_Float16*)alloc((size_t)ROWS_ * 512 * 2);
  _Float16* comb16 = t2_16;
  float*    h32    = (float*)alloc((size_t)ROWS_ * D_ * 4);
  _Float16* h16    = (_Float16*)alloc((size_t)ROWS_ * D_ * 2);
  float*    t1_32  = (float*)alloc((size_t)ROWS_ * D_ * 4);
  _Float16* t1_16  = (_Float16*)alloc((size_t)ROWS_ * D_ * 2);
  float*    mm     = (float*)alloc((size_t)BK_ * D_ * 4);
  float*    logit  = (float*)alloc((size_t)ROWS_ * 4);
  float*    psum   = (float*)alloc((size_t)2048 * D_ * 4);
  float*    psq    = (float*)alloc((size_t)2048 * D_ * 4);
  float*    stats  = (float*)alloc(512 * 4);
  float*    user32 = (float*)alloc((size_t)BK_ * D_ * 4);
  _Float16* user16 = (_Float16*)alloc((size_t)BK_ * D_ * 2);
  _Float16* ln16   = (_Float16*)alloc((size_t)BK_ * D_ * 2);
  float*    q32    = (float*)alloc((size_t)BK_ * D_ * 4);
  float*    k32    = (float*)alloc((size_t)BK_ * D_ * 4);
  float*    v32    = (float*)alloc((size_t)BK_ * D_ * 4);
  _Float16* o16    = (_Float16*)alloc((size_t)BK_ * D_ * 2);
  float*    t3_32  = (float*)alloc((size_t)BK_ * D_ * 4);

  auto cvt = [&](const float* src, int srcRows, int cols, int dstRows) -> _Float16* {
    _Float16* dst = (_Float16*)alloc((size_t)dstRows * cols * 2);
    size_t n = (size_t)dstRows * cols;
    int blocks = (int)((n + 255) / 256);
    if (blocks > 4096) blocks = 4096;
    cvt16_kernel<<<blocks, 256, 0, stream>>>(src, dst, srcRows, cols, n);
    return dst;
  };
  auto gemm = [&](const _Float16* A, int lda, const _Float16* W, int ldw, const float* bias,
                  float* C32, int ldc32, _Float16* C16, int ldc16,
                  int M, int Kd, int Nd, int relu) {
    dim3 g(Nd / 32, M / 64);
    gemm_wmma_kernel<<<g, 128, 0, stream>>>(A, lda, W, ldw, bias,
                                            C32, ldc32, C16, ldc16, M, Kd, Nd, relu);
  };
  auto addv = [&](float* X, const float* Y, size_t n) {
    add_kernel<<<2048, 256, 0, stream>>>(X, Y, n);
  };

  _Float16* fW1h = cvt(fW1, 262, D_, COMBW_);
  _Float16* fW2h = cvt(fW2, D_, D_, D_);
  _Float16 *gW[3][4];
  for (int i = 0; i < 3; ++i) {
    gW[i][0] = cvt(F(gbase[i] + 0), D_, D_, D_);
    gW[i][1] = cvt(F(gbase[i] + 2), D_, D_, D_);
    gW[i][2] = cvt(F(gbase[i] + 4), 2 * D_, D_, 2 * D_);
    gW[i][3] = cvt(F(gbase[i] + 6), D_, D_, D_);
  }
  _Float16* aW1h = cvt(aW1, D_, 128, D_);
  _Float16 *tW[2][6];
  for (int i = 0; i < 2; ++i) {
    tW[i][0] = cvt(F(tbase[i] + 0), D_, D_, D_);
    tW[i][1] = cvt(F(tbase[i] + 2), D_, D_, D_);
    tW[i][2] = cvt(F(tbase[i] + 4), D_, D_, D_);
    tW[i][3] = cvt(F(tbase[i] + 6), D_, D_, D_);
    tW[i][4] = cvt(F(tbase[i] + 8), D_, 2 * D_, D_);
    tW[i][5] = cvt(F(tbase[i] + 10), 2 * D_, D_, 2 * D_);
  }

  global_mmse_kernel<<<(B_ + 63) / 64, 64, 0, stream>>>(Hq, yq, sglob);
  per_ap_mmse_kernel<<<(B_ * L_ + 63) / 64, 64, 0, stream>>>(Hq, yq, s_init);

  FeatArgs fa = {s_init, s_hat, Hq, sglob, bitw, lsnr,
                 iW1, ib1, iW2, ib2, dW1, db1, dW2, db2,
                 gW1, gb1, gW2, gb2, cW1, cb1, cW2, cb2, comb16};
  features_kernel<<<ROWS_, 64, 0, stream>>>(fa);

  gemm(comb16, COMBW_, fW1h, D_, fb1, nullptr, 0, t1_16, D_, ROWS_, COMBW_, D_, 1);
  gemm(t1_16, D_, fW2h, D_, fb2, h32, D_, nullptr, 0, ROWS_, D_, D_, 0);
  bn_stat1_kernel<<<ROWS_ / 128, D_, 0, stream>>>(h32, psum, psq);
  bn_stat2_kernel<<<1, D_, 0, stream>>>(psum, psq, stats, ROWS_ / 128);
  bn_apply_kernel<<<4096, 256, 0, stream>>>(h32, stats, bng, bnb, h32, h16,
                                            (size_t)ROWS_ * D_);

  for (int i = 0; i < 3; ++i) {
    const float *mb1 = F(gbase[i] + 1), *mb2 = F(gbase[i] + 3);
    const float *ub1 = F(gbase[i] + 5), *ub2 = F(gbase[i] + 7);
    const float *lng = F(gbase[i] + 8), *lnb = F(gbase[i] + 9);
    gemm(h16, D_, gW[i][0], D_, mb1, nullptr, 0, t1_16, D_, ROWS_, D_, D_, 1);
    gemm(t1_16, D_, gW[i][1], D_, mb2, t1_32, D_, nullptr, 0, ROWS_, D_, D_, 0);
    mean_msg_kernel<<<(BK_ * D_ + 255) / 256, 256, 0, stream>>>(t1_32, mm);
    concat_upd_kernel<<<((size_t)ROWS_ * D_ + 255) / 256, 256, 0, stream>>>(h16, mm, t2_16);
    gemm(t2_16, 2 * D_, gW[i][2], D_, ub1, nullptr, 0, t1_16, D_, ROWS_, 2 * D_, D_, 1);
    gemm(t1_16, D_, gW[i][3], D_, ub2, t1_32, D_, nullptr, 0, ROWS_, D_, D_, 0);
    ln_kernel<<<ROWS_, D_, 0, stream>>>(h32, t1_32, lng, lnb, h32, h16);
  }

  gemm(h16, D_, aW1h, 128, ab1, nullptr, 0, t1_16, 128, ROWS_, D_, 128, 1);
  agg_logit_kernel<<<(ROWS_ + 255) / 256, 256, 0, stream>>>(t1_16, aW2, ab2, logit);
  attn_pool_kernel<<<BK_, D_, 0, stream>>>(h32, logit, user32, user16);

  for (int i = 0; i < 2; ++i) {
    const float *bq = F(tbase[i] + 1), *bk = F(tbase[i] + 3), *bv = F(tbase[i] + 5);
    const float *bo = F(tbase[i] + 7), *bf1 = F(tbase[i] + 9), *bf2 = F(tbase[i] + 11);
    const float *l1g = F(tbase[i] + 12), *l1b = F(tbase[i] + 13);
    const float *l2g = F(tbase[i] + 14), *l2b = F(tbase[i] + 15);
    ln_kernel<<<BK_, D_, 0, stream>>>(user32, nullptr, l1g, l1b, nullptr, ln16);
    gemm(ln16, D_, tW[i][0], D_, bq, q32, D_, nullptr, 0, BK_, D_, D_, 0);
    gemm(ln16, D_, tW[i][1], D_, bk, k32, D_, nullptr, 0, BK_, D_, D_, 0);
    gemm(ln16, D_, tW[i][2], D_, bv, v32, D_, nullptr, 0, BK_, D_, D_, 0);
    mha_attn_kernel<<<(B_ * NHEAD_ * K_ + 255) / 256, 256, 0, stream>>>(q32, k32, v32, o16);
    gemm(o16, D_, tW[i][3], D_, bo, t3_32, D_, nullptr, 0, BK_, D_, D_, 0);
    addv(user32, t3_32, (size_t)BK_ * D_);
    ln_kernel<<<BK_, D_, 0, stream>>>(user32, nullptr, l2g, l2b, nullptr, ln16);
    gemm(ln16, D_, tW[i][4], 2 * D_, bf1, nullptr, 0, t2_16, 2 * D_, BK_, D_, 2 * D_, 1);
    gemm(t2_16, 2 * D_, tW[i][5], D_, bf2, t3_32, D_, nullptr, 0, BK_, 2 * D_, D_, 0);
    addv(user32, t3_32, (size_t)BK_ * D_);
  }

  final_kernel<<<(BK_ + 255) / 256, 256, 0, stream>>>(user32, oW, ob, alpha, sglob, det);
}
